// WanT2VCrossAttentionGather_44976897524313
// MI455X (gfx1250) — compile-verified, matrix operand order checked by reading, not run on hardware
//
#include <hip/hip_runtime.h>
#include <hip/hip_bf16.h>
#include <math.h>

typedef __attribute__((ext_vector_type(16))) _Float16 v16h;
typedef __attribute__((ext_vector_type(8)))  _Float16 v8h;
typedef __attribute__((ext_vector_type(8)))  float    v8f;

#define DIM      5120
#define KV_DIM   1536
#define N_HEADS  40
#define HEAD_DIM 128
#define SEG      1560          // hw = 30*52
#define NBATCH   6             // F
#define L1TOK    (NBATCH*SEG)  // 9360
#define LATOK    96            // 6*16 audio tokens
#define EPSV     1e-6f

// ---------------------------------------------------------------------------
// Fragment loaders (wave32, 16-bit A/B layout per CDNA5 ISA 7.12.2):
//   lane = 0..31, r = lane&15, hi = lane>>4
//   halves 0..7  : K = hi*8 + 0..7        (contiguous 16B)
//   halves 8..15 : K = 16 + hi*8 + 0..7   (contiguous 16B)
// A fragment: row M = r of a 16x32 tile. B fragment: column N = r of a 32x16
// tile, i.e. row r of Bt when Bt is stored N x K row-major (K contiguous).
// ---------------------------------------------------------------------------
__device__ __forceinline__ v16h load_frag_f16(const _Float16* __restrict__ tile, int ld) {
    const int lane = threadIdx.x & 31;
    const int r = lane & 15, hi = lane >> 4;
    const _Float16* p = tile + (size_t)r * ld + hi * 8;
    v8h a0 = *reinterpret_cast<const v8h*>(p);
    v8h a1 = *reinterpret_cast<const v8h*>(p + 16);
    v16h out;
#pragma unroll
    for (int i = 0; i < 8; ++i) { out[i] = a0[i]; out[i + 8] = a1[i]; }
    return out;
}

// Same, but clamps the tile row index to [0, mlimit) (for ragged M edges).
__device__ __forceinline__ v16h load_frag_f16_clampM(const _Float16* __restrict__ tile,
                                                     int ld, int mlimit) {
    const int lane = threadIdx.x & 31;
    int r = lane & 15; const int hi = lane >> 4;
    if (r >= mlimit) r = 0;
    const _Float16* p = tile + (size_t)r * ld + hi * 8;
    v8h a0 = *reinterpret_cast<const v8h*>(p);
    v8h a1 = *reinterpret_cast<const v8h*>(p + 16);
    v16h out;
#pragma unroll
    for (int i = 0; i < 8; ++i) { out[i] = a0[i]; out[i + 8] = a1[i]; }
    return out;
}

// ---------------------------------------------------------------------------
// f32 -> f16 elementwise convert
// ---------------------------------------------------------------------------
__global__ void cvt_f32_to_f16(const float* __restrict__ in, _Float16* __restrict__ out,
                               size_t n) {
    size_t i = (size_t)blockIdx.x * blockDim.x + threadIdx.x;
    size_t stride = (size_t)gridDim.x * blockDim.x;
    for (; i < n; i += stride) out[i] = (_Float16)in[i];
}

// ---------------------------------------------------------------------------
// WMMA GEMM: C(MxN, f32) = A(MxK, f16, row-major) * B(NxK, f16, row-major)^T + bias
// Block: 256 threads = 8 waves (2 M x 4 N). Wave tile 64x64 = 4x4 16x16 tiles
// (16 WMMA per K-step vs 16 b128 loads -> 1:1, 4x operand reuse in registers).
// Block tile 128x256. N % 256 == 0, K % 32 == 0, M % 16 == 0 assumed.
// ---------------------------------------------------------------------------
__global__ __launch_bounds__(256) void gemm_f16_nt_wmma(
    const _Float16* __restrict__ A, const _Float16* __restrict__ B,
    const float* __restrict__ bias, float* __restrict__ C,
    int M, int N, int K) {
    const int wave = threadIdx.x >> 5;
    const int lane = threadIdx.x & 31;
    const int wm = wave >> 2;       // 0..1
    const int wn = wave & 3;        // 0..3
    const int m0 = blockIdx.y * 128 + wm * 64;
    const int n0 = blockIdx.x * 256 + wn * 64;

    v8f acc[4][4];
#pragma unroll
    for (int i = 0; i < 4; ++i)
#pragma unroll
        for (int j = 0; j < 4; ++j) acc[i][j] = (v8f){};

    const _Float16* Abase[4];
    bool mvalid[4];
#pragma unroll
    for (int i = 0; i < 4; ++i) {
        int mt = m0 + i * 16;
        mvalid[i] = (mt < M);                       // M % 16 == 0 -> full or empty
        Abase[i] = A + (mvalid[i] ? (size_t)mt * K : 0);
    }
    const _Float16* Bbase[4];
#pragma unroll
    for (int j = 0; j < 4; ++j) Bbase[j] = B + (size_t)(n0 + j * 16) * K;

    for (int k = 0; k < K; k += 32) {
        v16h a[4], b[4];
#pragma unroll
        for (int i = 0; i < 4; ++i) a[i] = load_frag_f16(Abase[i] + k, K);
#pragma unroll
        for (int j = 0; j < 4; ++j) b[j] = load_frag_f16(Bbase[j] + k, K);
#pragma unroll
        for (int i = 0; i < 4; ++i)
#pragma unroll
            for (int j = 0; j < 4; ++j)
                acc[i][j] = __builtin_amdgcn_wmma_f32_16x16x32_f16(
                    false, a[i], false, b[j], (short)0, acc[i][j], false, false);
    }

    // Epilogue: C/D layout — element (m,n): lane = (n&15) + 16*(m>=8), vgpr = m&7
    const int r = lane & 15, hi = lane >> 4;
#pragma unroll
    for (int i = 0; i < 4; ++i) {
        if (!mvalid[i]) continue;
#pragma unroll
        for (int j = 0; j < 4; ++j) {
            int n = n0 + j * 16 + r;
            float bv = bias[n];
#pragma unroll
            for (int v = 0; v < 8; ++v) {
                int m = m0 + i * 16 + v + hi * 8;
                C[(size_t)m * N + n] = acc[i][j][v] + bv;
            }
        }
    }
}

// ---------------------------------------------------------------------------
// Row-wise RMSNorm (over N cols) * g, f32 in -> f16 out. One block per row.
// ---------------------------------------------------------------------------
__global__ __launch_bounds__(256) void rmsnorm_to_f16(
    const float* __restrict__ X, const float* __restrict__ g,
    _Float16* __restrict__ Y, int N) {
    const int row = blockIdx.x;
    const float* x = X + (size_t)row * N;
    float ss = 0.f;
    for (int i = threadIdx.x; i < N; i += blockDim.x) { float v = x[i]; ss += v * v; }
#pragma unroll
    for (int off = 16; off; off >>= 1) ss += __shfl_xor(ss, off, 32);
    __shared__ float red[8];
    if ((threadIdx.x & 31) == 0) red[threadIdx.x >> 5] = ss;
    __syncthreads();
    if (threadIdx.x == 0) {
        float t = 0.f;
#pragma unroll
        for (int i = 0; i < 8; ++i) t += red[i];
        red[0] = t;
    }
    __syncthreads();
    const float scale = rsqrtf(red[0] / (float)N + EPSV);
    for (int i = threadIdx.x; i < N; i += blockDim.x)
        Y[(size_t)row * N + i] = (_Float16)(x[i] * scale * g[i]);
}

// ---------------------------------------------------------------------------
// Cross attention, 16 keys per (batch-segment, head).
// Grid: (NBATCH, N_HEADS), block 256 = 8 waves. Each wave owns successive
// 16-query tiles. K/V fragments are loaded once per wave.
// ---------------------------------------------------------------------------
__global__ __launch_bounds__(256) void cross_attn_wmma(
    const _Float16* __restrict__ Q,   // (NBATCH*SEG) x DIM, rmsnormed
    const _Float16* __restrict__ Kk,  // (NBATCH*16)  x DIM, rmsnormed
    const _Float16* __restrict__ V,   // (NBATCH*16)  x DIM
    _Float16* __restrict__ O) {       // (NBATCH*SEG) x DIM
    const int b = blockIdx.x;
    const int h = blockIdx.y;
    const int wave = threadIdx.x >> 5;
    const int lane = threadIdx.x & 31;
    const int r = lane & 15, hi = lane >> 4;
    const int ld = DIM;
    const float sm_scale = 0.08838834764831845f;  // 1/sqrt(128)

    __shared__ float             s_sm[8][16][17];  // per-wave score slab
    __shared__ alignas(16) _Float16 s_p[8][16][16]; // per-wave normalized P (f16)

    // K fragments: B-operand of S = Q*K^T. 4 chunks over head_dim (K=32 each).
    const _Float16* Kb = Kk + (size_t)(b * 16) * ld + h * HEAD_DIM;
    v16h kf[4];
#pragma unroll
    for (int c = 0; c < 4; ++c) kf[c] = load_frag_f16(Kb + c * 32, ld);

    // V fragments: B-operand of O = P*V, one per 16-wide d-tile. Only keys
    // 0..15 exist -> halves 8..15 (K=16..31) are explicit zeros.
    const _Float16* Vb = V + (size_t)(b * 16) * ld + h * HEAD_DIM;
    v16h vf[8];
#pragma unroll
    for (int t = 0; t < 8; ++t) {
        v16h f;
#pragma unroll
        for (int j = 0; j < 8; ++j) {
            f[j]     = Vb[(size_t)(hi * 8 + j) * ld + t * 16 + r];
            f[j + 8] = (_Float16)0.f;
        }
        vf[t] = f;
    }

    const int ntiles = (SEG + 15) / 16;  // 98 (last tile has 8 rows)
    for (int tile = wave; tile < ntiles; tile += 8) {
        const int q0 = tile * 16;
        const int mlim = SEG - q0 < 16 ? SEG - q0 : 16;
        const _Float16* Qb = Q + (size_t)(b * SEG + q0) * ld + h * HEAD_DIM;

        // S = Q * K^T  (16x16 f32)
        v8f s = (v8f){};
#pragma unroll
        for (int c = 0; c < 4; ++c) {
            v16h a = load_frag_f16_clampM(Qb + c * 32, ld, mlim);
            s = __builtin_amdgcn_wmma_f32_16x16x32_f16(
                false, a, false, kf[c], (short)0, s, false, false);
        }
        // scatter scores: element (m = v + hi*8, n = r)
#pragma unroll
        for (int v = 0; v < 8; ++v) s_sm[wave][v + hi * 8][r] = s[v];

        // softmax over the 16 keys (lanes 0..15, one row each); wave-private
        if (lane < 16) {
            float mx = -3.0e38f;
#pragma unroll
            for (int n = 0; n < 16; ++n) {
                float t = s_sm[wave][lane][n] * sm_scale;
                mx = t > mx ? t : mx;
            }
            float sum = 0.f;
            float e[16];
#pragma unroll
            for (int n = 0; n < 16; ++n) {
                e[n] = __expf(s_sm[wave][lane][n] * sm_scale - mx);
                sum += e[n];
            }
            float inv = 1.f / sum;
#pragma unroll
            for (int n = 0; n < 16; ++n)
                s_p[wave][lane][n] = (_Float16)(e[n] * inv);
        }

        // Rebuild P as an A-fragment (K = key index, padded 16..31 with zeros)
        v8h plo = *reinterpret_cast<const v8h*>(&s_p[wave][r][hi * 8]);
        v16h pf;
#pragma unroll
        for (int j = 0; j < 8; ++j) { pf[j] = plo[j]; pf[j + 8] = (_Float16)0.f; }

        // O tiles = P * V, store f16
        _Float16* Ob = O + (size_t)(b * SEG + q0) * ld + h * HEAD_DIM;
#pragma unroll
        for (int t = 0; t < 8; ++t) {
            v8f o = __builtin_amdgcn_wmma_f32_16x16x32_f16(
                false, pf, false, vf[t], (short)0, (v8f){}, false, false);
#pragma unroll
            for (int v = 0; v < 8; ++v) {
                int m = v + hi * 8;
                if (m < mlim)
                    Ob[(size_t)m * ld + t * 16 + r] = (_Float16)o[v];
            }
        }
    }
}

// ---------------------------------------------------------------------------
extern "C" void kernel_launch(void* const* d_in, const int* in_sizes, int n_in,
                              void* d_out, int out_size, void* d_ws, size_t ws_size,
                              hipStream_t stream) {
    const float* x   = (const float*)d_in[0];
    const float* ctx = (const float*)d_in[1];
    const float* Wq  = (const float*)d_in[6];
    const float* bq  = (const float*)d_in[7];
    const float* Wk  = (const float*)d_in[8];
    const float* bk  = (const float*)d_in[9];
    const float* Wv  = (const float*)d_in[10];
    const float* bv  = (const float*)d_in[11];
    const float* Wo  = (const float*)d_in[12];
    const float* bo  = (const float*)d_in[13];
    const float* gq  = (const float*)d_in[14];
    const float* gk  = (const float*)d_in[15];
    float* out = (float*)d_out;

    // ---- workspace carve (256B aligned) ----
    char* w = (char*)d_ws;
    auto carve = [&](size_t bytes) {
        void* p = (void*)w;
        w += (bytes + 255) & ~(size_t)255;
        return p;
    };
    const size_t nX  = (size_t)L1TOK * DIM;       // 9360*5120
    const size_t nCx = (size_t)LATOK * KV_DIM;    // 96*1536
    const size_t nWq = (size_t)DIM * DIM;
    const size_t nWk = (size_t)DIM * KV_DIM;
    const size_t nKV = (size_t)LATOK * DIM;       // 96*5120

    _Float16* x_h   = (_Float16*)carve(nX * 2);
    _Float16* ctx_h = (_Float16*)carve(nCx * 2);
    _Float16* wq_h  = (_Float16*)carve(nWq * 2);
    _Float16* wk_h  = (_Float16*)carve(nWk * 2);
    _Float16* wv_h  = (_Float16*)carve(nWk * 2);
    _Float16* wo_h  = (_Float16*)carve(nWq * 2);
    float*    qfull = (float*)   carve(nX * 4);   // f32 Q projection (dead after rmsnorm)
    _Float16* q_h   = (_Float16*)carve(nX * 2);
    float*    kfull = (float*)   carve(nKV * 4);
    _Float16* k_h   = (_Float16*)carve(nKV * 2);
    float*    vfull = (float*)   carve(nKV * 4);
    _Float16* v_h   = (_Float16*)carve(nKV * 2);
    _Float16* o_h   = (_Float16*)qfull;           // alias dead buffer

    // ---- f32 -> f16 converts ----
    cvt_f32_to_f16<<<2048, 256, 0, stream>>>(x,   x_h,   nX);
    cvt_f32_to_f16<<<256,  256, 0, stream>>>(ctx, ctx_h, nCx);
    cvt_f32_to_f16<<<2048, 256, 0, stream>>>(Wq,  wq_h,  nWq);
    cvt_f32_to_f16<<<1024, 256, 0, stream>>>(Wk,  wk_h,  nWk);
    cvt_f32_to_f16<<<1024, 256, 0, stream>>>(Wv,  wv_h,  nWk);
    cvt_f32_to_f16<<<2048, 256, 0, stream>>>(Wo,  wo_h,  nWq);

    // ---- Q = rmsnorm(x @ Wq^T + bq) * gq ----
    {
        dim3 grid(DIM / 256, (L1TOK + 127) / 128);
        gemm_f16_nt_wmma<<<grid, 256, 0, stream>>>(x_h, wq_h, bq, qfull,
                                                   L1TOK, DIM, DIM);
        rmsnorm_to_f16<<<L1TOK, 256, 0, stream>>>(qfull, gq, q_h, DIM);
    }
    // ---- K = rmsnorm(ctx @ Wk^T + bk) * gk ----
    {
        dim3 grid(DIM / 256, 1);
        gemm_f16_nt_wmma<<<grid, 256, 0, stream>>>(ctx_h, wk_h, bk, kfull,
                                                   LATOK, DIM, KV_DIM);
        rmsnorm_to_f16<<<LATOK, 256, 0, stream>>>(kfull, gk, k_h, DIM);
    }
    // ---- V = ctx @ Wv^T + bv ----
    {
        dim3 grid(DIM / 256, 1);
        gemm_f16_nt_wmma<<<grid, 256, 0, stream>>>(ctx_h, wv_h, bv, vfull,
                                                   LATOK, DIM, KV_DIM);
        cvt_f32_to_f16<<<256, 256, 0, stream>>>(vfull, v_h, nKV);
    }
    // ---- attention ----
    {
        dim3 grid(NBATCH, N_HEADS);
        cross_attn_wmma<<<grid, 256, 0, stream>>>(q_h, k_h, v_h, o_h);
    }
    // ---- out = o @ Wo^T + bo (f32 to d_out) ----
    {
        dim3 grid(DIM / 256, (L1TOK + 127) / 128);
        gemm_f16_nt_wmma<<<grid, 256, 0, stream>>>(o_h, wo_h, bo, out,
                                                   L1TOK, DIM, DIM);
    }
}
